// GeneTreeGIN_90357521973606
// MI455X (gfx1250) — matrix operand order repaired and checked by reading, not compiled, not run on hardware
//
#include <hip/hip_runtime.h>
#include <hip/hip_bf16.h>
#include <math.h>

// ---------------------------------------------------------------------------
// GeneTreeGIN on MI455X (gfx1250, wave32).
//  - All dense math through v_wmma_f32_16x16x32_f16 (exact for the integer
//    count matrices; adequate precision for the D=64 MLP / pooled embeddings).
//  - Fused MLP tile kernel: one 16-row tile per wave, staged in LDS.
//  - Per-edge einsum: pooled-embedding panel (28.7MB f16) lives in L2;
//    gfx1250 async global->LDS engine (ASYNCcnt) double-buffers each 28KB
//    panel in LDS so the next panel's fetch overlaps the current 28 WMMAs.
// ---------------------------------------------------------------------------

typedef _Float16 h16;
typedef __attribute__((ext_vector_type(16))) _Float16 v16h;
typedef __attribute__((ext_vector_type(8)))  _Float16 v8h;
typedef __attribute__((ext_vector_type(8)))  float    v8f;

#define NSP  200
#define NGT  1000
#define NND  400
#define NTE  399
#define NED  400
#define DD   64
#define SPAD 224          // 7 * 32  (K padding for species dim)
#define GPAD 1008         // 63 * 16 (N padding for gene-tree dim)
#define OUTC 138          // 2*D + 10
#define NTOT (NGT * NND)  // 400000 graph nodes

#if defined(__has_builtin)
#if __has_builtin(__builtin_amdgcn_global_load_async_to_lds_b128) && \
    __has_builtin(__builtin_amdgcn_s_wait_asynccnt)
#define GTG_HAVE_ASYNC 1
#endif
#endif
#ifndef GTG_HAVE_ASYNC
#define GTG_HAVE_ASYNC 0
#endif

// ---------------- WMMA fragment helpers (ISA 7.12.2 layouts) ----------------
// A (16x32 f16): lane m=l&15; halves 0..7 -> K = kb + hi*8 .. +7,
//                halves 8..15 -> K = kb + 16 + hi*8 .. +7   (hi = l>>4)
__device__ __forceinline__ v16h afrag(const h16* base, int stride, int kb, int lane) {
  int m = lane & 15, hi = (lane >> 4) & 1;
  const h16* p = base + (size_t)m * stride + kb + hi * 8;
  union { v16h v; v8h h[2]; } u;
  u.h[0] = *(const v8h*)(p);
  u.h[1] = *(const v8h*)(p + 16);
  return u.v;
}
// B (32x16 f16), stored as [n][k] row-major: lane n=l&15; halves 0..7 -> K =
// kb + hi*8 .. +7, halves 8..15 -> K = kb + 16 + hi*8 .. +7 (same as A).
__device__ __forceinline__ v16h bfrag(const h16* base, int stride, int kb, int lane) {
  int n = lane & 15, hi = (lane >> 4) & 1;
  const h16* p = base + (size_t)n * stride + kb + hi * 8;
  union { v16h v; v8h h[2]; } u;
  u.h[0] = *(const v8h*)(p);
  u.h[1] = *(const v8h*)(p + 16);
  return u.v;
}
__device__ __forceinline__ v8f wmma16(v16h a, v16h b, v8f c) {
  return __builtin_amdgcn_wmma_f32_16x16x32_f16(false, a, false, b, (short)0, c,
                                                false, false);
}
// sum across a 16-lane half (wave32 butterfly)
__device__ __forceinline__ float red16(float v) {
  v += __shfl_xor(v, 1); v += __shfl_xor(v, 2);
  v += __shfl_xor(v, 4); v += __shfl_xor(v, 8);
  return v;
}

// ------------------------------- tiny kernels -------------------------------
__global__ void gtg_w2h(const float* W1, const float* W2, h16* W1h, h16* W2h) {
  int i = blockIdx.x * blockDim.x + threadIdx.x;
  if (i < 2 * DD * DD) { W1h[i] = (h16)W1[i]; W2h[i] = (h16)W2[i]; }
}

__global__ void gtg_count(const int* sid, const int* lm, float* spc) {
  int i = blockIdx.x * blockDim.x + threadIdx.x;
  if (i >= NTOT) return;
  int s = sid[i];
  if (lm[i] != 0 && s >= 0) atomicAdd(&spc[(i / NND) * NSP + s], 1.0f);
}

__global__ void gtg_masks(const float* spc, h16* vfh, h16* cnth, h16* duph) {
  int i = blockIdx.x * blockDim.x + threadIdx.x;
  if (i >= GPAD * SPAD) return;
  int g = i / SPAD, s = i % SPAD;
  float c = (g < NGT && s < NSP) ? spc[g * NSP + s] : 0.0f;
  vfh[i]  = (h16)(c > 0.0f ? 1.0f : 0.0f);
  cnth[i] = (h16)c;                          // sp_count * vf == sp_count
  duph[i] = (h16)(c > 1.0f ? 1.0f : 0.0f);
}

__global__ void gtg_totals(const float* spc, float* tv, float* tc, float* td) {
  int g = blockIdx.x * blockDim.x + threadIdx.x;
  if (g >= GPAD) return;
  float v = 0.f, c = 0.f, d = 0.f;
  if (g < NGT)
    for (int s = 0; s < NSP; ++s) {
      float x = spc[g * NSP + s];
      v += (x > 0.f) ? 1.f : 0.f; c += x; d += (x > 1.f) ? 1.f : 0.f;
    }
  tv[g] = v; tc[g] = c; td[g] = d;
}

__global__ void gtg_cm(const int* cmi, h16* cmh) {
  int i = blockIdx.x * blockDim.x + threadIdx.x;
  if (i >= NED * SPAD) return;
  int e = i / SPAD, s = i % SPAD;
  cmh[i] = (h16)((s < NSP && cmi[e * NSP + s] != 0) ? 1.0f : 0.0f);
}

__global__ void gtg_cmrow(const int* cmi, float* rs) {
  int e = blockIdx.x * blockDim.x + threadIdx.x;
  if (e >= NED) return;
  int n = 0;
  for (int s = 0; s < NSP; ++s) n += (cmi[e * NSP + s] != 0);
  rs[e] = (float)n;
}

__global__ void gtg_initx(const int* sid, const float* emb, float* x) {
  int i = blockIdx.x * blockDim.x + threadIdx.x;
  if (i >= NTOT * DD) return;
  int node = i >> 6, d = i & 63;
  int s = sid[node];
  x[i] = emb[(s < 0 ? NSP : s) * DD + d];
}

__global__ void gtg_scatter(const int* ei, const float* x, float* agg) {
  int t = blockIdx.x * blockDim.x + threadIdx.x;
  if (t >= NGT * NTE * 16) return;
  int e = t >> 4, ch = t & 15;
  int g = e / NTE, k = e % NTE;
  int src = g * NND + ei[g * 2 * NTE + k];
  int dst = g * NND + ei[g * 2 * NTE + NTE + k];
  int d0 = ch * 4;
#pragma unroll
  for (int j = 0; j < 4; ++j)
    atomicAdd(&agg[(size_t)dst * DD + d0 + j], x[(size_t)src * DD + d0 + j]);
}

// ------------------ fused GIN layer: MLP + residual + LN --------------------
__global__ void __launch_bounds__(128)
gtg_mlp(float* x, const float* agg, const h16* W1h, const float* b1,
        const h16* W2h, const float* b2, const float* eps_p,
        const float* gamma, const float* beta) {
  __shared__ __align__(16) float xin [4][16][DD];
  __shared__ __align__(16) h16   xpre[4][16][DD];
  __shared__ __align__(16) h16   h1s [4][16][DD];
  __shared__ __align__(16) float xnew[4][16][DD];
  int wave = threadIdx.x >> 5, lane = threadIdx.x & 31;
  int ni = lane & 15, hi = lane >> 4;
  int tile = blockIdx.x * 4 + wave;           // 25000 tiles total
  int row0 = tile * 16;
  float ev = 1.0f + eps_p[0];

  // stage 1: load x/agg, pre-activation into f16
  for (int i = lane; i < 16 * DD; i += 32) {
    int r = i >> 6, c = i & 63;
    float xv = x[(size_t)(row0 + r) * DD + c];
    float av = agg[(size_t)(row0 + r) * DD + c];
    xin[wave][r][c]  = xv;
    xpre[wave][r][c] = (h16)(ev * xv + av);
  }
  __syncthreads();

  // stage 2: h1 = relu(xpre @ W1^T + b1)   (B[k,n] = W1[n,k] -> [n][k] storage)
  const h16* xpw = &xpre[wave][0][0];
#pragma unroll
  for (int nt = 0; nt < 4; ++nt) {
    v8f acc = {};
#pragma unroll
    for (int kb = 0; kb < DD; kb += 32)
      acc = wmma16(afrag(xpw, DD, kb, lane),
                   bfrag(W1h + nt * 16 * DD, DD, kb, lane), acc);
    float bb = b1[nt * 16 + ni];
#pragma unroll
    for (int r = 0; r < 8; ++r) {
      float v = acc[r] + bb;
      h1s[wave][r + 8 * hi][nt * 16 + ni] = (h16)(v > 0.f ? v : 0.f);
    }
  }
  __syncthreads();

  // stage 3: h2 = h1 @ W2^T + b2 ; xnew = x + h2
  const h16* h1w = &h1s[wave][0][0];
#pragma unroll
  for (int nt = 0; nt < 4; ++nt) {
    v8f acc = {};
#pragma unroll
    for (int kb = 0; kb < DD; kb += 32)
      acc = wmma16(afrag(h1w, DD, kb, lane),
                   bfrag(W2h + nt * 16 * DD, DD, kb, lane), acc);
    float bb = b2[nt * 16 + ni];
#pragma unroll
    for (int r = 0; r < 8; ++r) {
      int m = r + 8 * hi, c = nt * 16 + ni;
      xnew[wave][m][c] = xin[wave][m][c] + acc[r] + bb;
    }
  }
  __syncthreads();

  // stage 4: LayerNorm over D=64 (2 lanes per row)
  int row = lane >> 1, c0 = (lane & 1) * 32;
  float s = 0.f, q = 0.f;
  for (int j = 0; j < 32; ++j) {
    float v = xnew[wave][row][c0 + j];
    s += v; q += v * v;
  }
  s += __shfl_xor(s, 1); q += __shfl_xor(q, 1);
  float mu = s * (1.f / 64.f);
  float rs = rsqrtf(q * (1.f / 64.f) - mu * mu + 1e-5f);
  for (int j = 0; j < 32; ++j) {
    int c = c0 + j;
    x[(size_t)(row0 + row) * DD + c] =
        (xnew[wave][row][c] - mu) * rs * gamma[c] + beta[c];
  }
}

// --------------------------- contrast (WMMA) --------------------------------
// one wave per 16-edge tile; 3 GEMMs [16e x S] x [S x 16g] per g-tile.
__global__ void __launch_bounds__(32)
gtg_contrast(const h16* cmh, const h16* vfh, const h16* cnth, const h16* duph,
             const float* tv, const float* tc, const float* td,
             const float* cmrow, float* rcpnb, float* out) {
  int et = blockIdx.x, lane = threadIdx.x, hi = lane >> 4;
  const h16* A = cmh + (size_t)et * 16 * SPAD;
  v16h afr[7];
#pragma unroll
  for (int kt = 0; kt < 7; ++kt) afr[kt] = afrag(A, SPAD, kt * 32, lane);

  float sA[5][8], sQ[5][8], cn[8];
#pragma unroll
  for (int f = 0; f < 5; ++f)
#pragma unroll
    for (int r = 0; r < 8; ++r) { sA[f][r] = 0.f; sQ[f][r] = 0.f; }
#pragma unroll
  for (int r = 0; r < 8; ++r) cn[r] = 0.f;

  for (int gt = 0; gt < GPAD / 16; ++gt) {
    v8f nb = {}, sb = {}, db = {};
#pragma unroll
    for (int kt = 0; kt < 7; ++kt) {
      int kb = kt * 32;
      nb = wmma16(afr[kt], bfrag(vfh  + (size_t)gt * 16 * SPAD, SPAD, kb, lane), nb);
      sb = wmma16(afr[kt], bfrag(cnth + (size_t)gt * 16 * SPAD, SPAD, kb, lane), sb);
      db = wmma16(afr[kt], bfrag(duph + (size_t)gt * 16 * SPAD, SPAD, kb, lane), db);
    }
    int g = gt * 16 + (lane & 15);
    float tvg = tv[g], tcg = tc[g], tdg = td[g];
#pragma unroll
    for (int r = 0; r < 8; ++r) {
      float nbv = nb[r], sbv = sb[r], dbv = db[r];
      float nov = tvg - nbv, sov = tcg - sbv, dov = tdg - dbv;
      float avgb = sbv / fmaxf(nbv, 1.f);
      float avgo = sov / fmaxf(nov, 1.f);
      float cr   = avgb / fmaxf(avgo, 0.1f);
      float fdb  = dbv / fmaxf(nbv, 1.f);
      float fdo  = dov / fmaxf(nov, 1.f);
      float fdd  = fdb - fdo;
      float hf = (nbv > 0.f && nov > 0.f) ? 1.f : 0.f;
      sA[0][r] += avgb * hf; sQ[0][r] += avgb * avgb * hf;
      sA[1][r] += cr   * hf; sQ[1][r] += cr   * cr   * hf;
      sA[2][r] += fdb  * hf; sQ[2][r] += fdb  * fdb  * hf;
      sA[3][r] += fdo  * hf; sQ[3][r] += fdo  * fdo  * hf;
      sA[4][r] += fdd  * hf; sQ[4][r] += fdd  * fdd  * hf;
      cn[r] += hf;
      int e = et * 16 + r + 8 * hi;
      rcpnb[(size_t)e * GPAD + g] = nbv > 0.f ? 1.f / nbv : 0.f;
    }
  }
  // reduce over the 16 gene-tree columns held by this half-wave
#pragma unroll
  for (int r = 0; r < 8; ++r) {
#pragma unroll
    for (int f = 0; f < 5; ++f) { sA[f][r] = red16(sA[f][r]); sQ[f][r] = red16(sQ[f][r]); }
    cn[r] = red16(cn[r]);
  }
  if ((lane & 15) == 0) {
#pragma unroll
    for (int r = 0; r < 8; ++r) {
      int e = et * 16 + r + 8 * hi;
      float n = cn[r];
      bool ok = (cmrow[e] > 0.f) && (cmrow[e] < (float)NSP) && (n > 0.f);
      float* o = out + (size_t)e * OUTC + 2 * DD;
#pragma unroll
      for (int f = 0; f < 5; ++f) {
        float m  = sA[f][r] / fmaxf(n, 1.f);
        float vr = fmaxf(sQ[f][r] - m * sA[f][r], 0.f) / fmaxf(n - 1.f, 1.f);
        float sd = (n > 1.f) ? sqrtf(vr) : 0.f;
        o[2 * f]     = ok ? m  : 0.f;
        o[2 * f + 1] = ok ? sd : 0.f;
      }
    }
  }
}

// ------------------------------ pooling -------------------------------------
__global__ void gtg_pool(const int* sid, const int* lm, const float* x, float* ps) {
  int t = blockIdx.x * blockDim.x + threadIdx.x;
  if (t >= NTOT * 16) return;
  int node = t >> 4, ch = t & 15;
  int s = sid[node];
  if (lm[node] == 0 || s < 0) return;
  int g = node / NND, d0 = ch * 4;
  size_t dst = ((size_t)g * NSP + s) * DD + d0;
  size_t src = (size_t)node * DD + d0;
#pragma unroll
  for (int j = 0; j < 4; ++j) atomicAdd(&ps[dst + j], x[src + j]);
}

// pooledT[g][d][s] f16 (k-major panels for B fragments)
__global__ void gtg_poolT(const float* ps, const float* spc, h16* pT) {
  int i = blockIdx.x * blockDim.x + threadIdx.x;
  if (i >= NGT * DD * SPAD) return;
  int g = i / (DD * SPAD), r = i % (DD * SPAD);
  int d = r / SPAD, s = r % SPAD;
  float v = 0.f;
  if (s < NSP) {
    float c = spc[g * NSP + s];
    v = ps[((size_t)g * NSP + s) * DD + d] / fmaxf(c, 1.f);
  }
  pT[i] = (h16)v;
}

// ----------------- per-edge clade einsum + stats (WMMA) ----------------------
#if GTG_HAVE_ASYNC
// Builtin signature (from hipcc diagnostic): param pointees are gcc-style
// 4-int vectors qualified with target AS1 (global) / AS3 (LDS).
typedef int v4i_vs __attribute__((vector_size(16)));
typedef v4i_vs __attribute__((address_space(1))) as1_v4i;
typedef v4i_vs __attribute__((address_space(3))) as3_v4i;

// cooperative async copy of one 28672B panel: 56 x (32 lanes x 16B), ASYNCcnt.
// AS pointers built via integer casts: flat==AS1 numerically for global; for
// LDS the flat aperture keeps the LDS byte offset in addr[31:0] (ISA 10.2).
__device__ __forceinline__ void gtg_panel_async(const h16* gsrc, h16* ldst, int lane) {
  uintptr_t g = (uintptr_t)gsrc + (uintptr_t)(lane * 16);
  unsigned  l = (unsigned)(uintptr_t)ldst + (unsigned)(lane * 16);
#pragma unroll
  for (int i = 0; i < 56; ++i)
    __builtin_amdgcn_global_load_async_to_lds_b128(
        (as1_v4i*)(g + (uintptr_t)i * 512u),
        (as3_v4i*)(size_t)(l + (unsigned)i * 512u), 0, 0);
}
#endif

__global__ void __launch_bounds__(32)
gtg_edge(const h16* cmh, const h16* pT, const float* rcpnb,
         const float* cmrow, float* out) {
  int et = blockIdx.x, lane = threadIdx.x, hi = lane >> 4;
  const h16* A = cmh + (size_t)et * 16 * SPAD;
  v16h afr[7];
#pragma unroll
  for (int kt = 0; kt < 7; ++kt) afr[kt] = afrag(A, SPAD, kt * 32, lane);

  float sm[4][8], sq[4][8], n2[8];
#pragma unroll
  for (int nt = 0; nt < 4; ++nt)
#pragma unroll
    for (int r = 0; r < 8; ++r) { sm[nt][r] = 0.f; sq[nt][r] = 0.f; }
#pragma unroll
  for (int r = 0; r < 8; ++r) n2[r] = 0.f;

#if GTG_HAVE_ASYNC
  // double-buffered LDS panels, async-engine filled, ds_load_b128 consumed
  __shared__ __align__(16) h16 bbuf[2][DD * SPAD];   // 2 x 28672 B
  gtg_panel_async(pT, bbuf[0], lane);
#endif

  for (int g = 0; g < NGT; ++g) {
#if GTG_HAVE_ASYNC
    if (g + 1 < NGT) {
      // WAR: all ds_loads of buffer (g+1)&1 (from iteration g-1) must retire
      // before the async engine overwrites it.
      asm volatile("s_wait_dscnt 0" ::: "memory");
      gtg_panel_async(pT + (size_t)(g + 1) * DD * SPAD, bbuf[(g + 1) & 1], lane);
      // async loads complete in order: <=56 outstanding => panel g landed,
      // panel g+1's 56 copies remain in flight behind the WMMAs below.
      __builtin_amdgcn_s_wait_asynccnt(56);
    } else {
      __builtin_amdgcn_s_wait_asynccnt(0);
    }
    const h16* B = bbuf[g & 1];
#else
    const h16* B = pT + (size_t)g * DD * SPAD;
    if (g + 1 < NGT)  // pull next 28KB panel toward L2/L0 (global_prefetch_b8)
      __builtin_prefetch(pT + (size_t)(g + 1) * DD * SPAD + lane * 448, 0, 1);
#endif
    float rl = (lane < 16)
                   ? rcpnb[(size_t)(et * 16 + lane) * GPAD + g] : 0.f;
    v8f acc[4];
#pragma unroll
    for (int nt = 0; nt < 4; ++nt) {
      v8f a = {};
#pragma unroll
      for (int kt = 0; kt < 7; ++kt)
        a = wmma16(afr[kt], bfrag(B + nt * 16 * SPAD, SPAD, kt * 32, lane), a);
      acc[nt] = a;
    }
#pragma unroll
    for (int r = 0; r < 8; ++r) {
      float rv = __shfl(rl, r + 8 * hi);      // 1/n_b for this lane's edge row
      n2[r] += (rv > 0.f) ? 1.f : 0.f;
#pragma unroll
      for (int nt = 0; nt < 4; ++nt) {
        float gm = acc[nt][r] * rv;           // gt_mean * h2
        sm[nt][r] += gm; sq[nt][r] += gm * gm;
      }
    }
  }
#pragma unroll
  for (int r = 0; r < 8; ++r) {
    int e = et * 16 + r + 8 * hi;
    float n = n2[r];
    bool ok = (cmrow[e] > 0.f) && (n > 0.f);
#pragma unroll
    for (int nt = 0; nt < 4; ++nt) {
      int d = nt * 16 + (lane & 15);
      float m  = sm[nt][r] / fmaxf(n, 1.f);
      float vr = fmaxf(sq[nt][r] - m * sm[nt][r], 0.f) / fmaxf(n - 1.f, 1.f);
      float sd = (n > 1.f) ? sqrtf(vr) : 0.f;
      out[(size_t)e * OUTC + d]      = ok ? m  : 0.f;
      out[(size_t)e * OUTC + DD + d] = ok ? sd : 0.f;
    }
  }
}

// ------------------------------- launcher -----------------------------------
extern "C" void kernel_launch(void* const* d_in, const int* in_sizes, int n_in,
                              void* d_out, int out_size, void* d_ws, size_t ws_size,
                              hipStream_t stream) {
  const int*   sid  = (const int*)d_in[0];
  const int*   lm   = (const int*)d_in[1];
  const int*   ei   = (const int*)d_in[2];
  const int*   cmi  = (const int*)d_in[3];
  const float* emb  = (const float*)d_in[4];
  const float* W1   = (const float*)d_in[5];
  const float* b1   = (const float*)d_in[6];
  const float* W2   = (const float*)d_in[7];
  const float* b2   = (const float*)d_in[8];
  const float* eps  = (const float*)d_in[9];
  const float* gam  = (const float*)d_in[10];
  const float* bet  = (const float*)d_in[11];
  float* out = (float*)d_out;

  // ---- workspace carving (requires ~290MB; assumed provisioned) ----
  char* w = (char*)d_ws;
  auto carve = [&](size_t bytes) {
    void* p = (void*)w;
    w += (bytes + 255) & ~(size_t)255;
    return p;
  };
  float* x    = (float*)carve((size_t)NTOT * DD * 4);
  float* agg  = (float*)carve((size_t)NTOT * DD * 4);
  float* ps   = (float*)carve((size_t)NGT * NSP * DD * 4);
  h16*   pT   = (h16*)  carve((size_t)NGT * DD * SPAD * 2);
  float* spc  = (float*)carve((size_t)NGT * NSP * 4);
  h16*   vfh  = (h16*)  carve((size_t)GPAD * SPAD * 2);
  h16*   cnth = (h16*)  carve((size_t)GPAD * SPAD * 2);
  h16*   duph = (h16*)  carve((size_t)GPAD * SPAD * 2);
  h16*   cmh  = (h16*)  carve((size_t)NED * SPAD * 2);
  float* tv   = (float*)carve(GPAD * 4);
  float* tc   = (float*)carve(GPAD * 4);
  float* td   = (float*)carve(GPAD * 4);
  float* cmr  = (float*)carve(NED * 4);
  float* rnb  = (float*)carve((size_t)NED * GPAD * 4);
  h16*   W1h  = (h16*)  carve((size_t)2 * DD * DD * 2);
  h16*   W2h  = (h16*)  carve((size_t)2 * DD * DD * 2);

  const int T = 256;
  // weights -> f16
  gtg_w2h<<<(2 * DD * DD + T - 1) / T, T, 0, stream>>>(W1, W2, W1h, W2h);
  // copy counts
  (void)hipMemsetAsync(spc, 0, (size_t)NGT * NSP * 4, stream);
  gtg_count<<<(NTOT + T - 1) / T, T, 0, stream>>>(sid, lm, spc);
  // mask/count/dup f16 tables + per-g totals
  gtg_masks<<<(GPAD * SPAD + T - 1) / T, T, 0, stream>>>(spc, vfh, cnth, duph);
  gtg_totals<<<(GPAD + T - 1) / T, T, 0, stream>>>(spc, tv, tc, td);
  gtg_cm<<<(NED * SPAD + T - 1) / T, T, 0, stream>>>(cmi, cmh);
  gtg_cmrow<<<(NED + T - 1) / T, T, 0, stream>>>(cmi, cmr);
  // contrast features (also produces 1/n_b table for the einsum stage)
  gtg_contrast<<<NED / 16, 32, 0, stream>>>(cmh, vfh, cnth, duph,
                                            tv, tc, td, cmr, rnb, out);
  // GIN
  gtg_initx<<<(NTOT * DD + T - 1) / T, T, 0, stream>>>(sid, emb, x);
  for (int l = 0; l < 2; ++l) {
    (void)hipMemsetAsync(agg, 0, (size_t)NTOT * DD * 4, stream);
    gtg_scatter<<<(NGT * NTE * 16 + T - 1) / T, T, 0, stream>>>(ei, x, agg);
    gtg_mlp<<<NTOT / 16 / 4, 128, 0, stream>>>(
        x, agg, W1h + (size_t)l * DD * DD, b1 + l * DD,
        W2h + (size_t)l * DD * DD, b2 + l * DD, eps + l,
        gam + l * DD, bet + l * DD);
  }
  // per-(gt,species) pooling
  (void)hipMemsetAsync(ps, 0, (size_t)NGT * NSP * DD * 4, stream);
  gtg_pool<<<(NTOT * 16 + T - 1) / T, T, 0, stream>>>(sid, lm, x, ps);
  gtg_poolT<<<((size_t)NGT * DD * SPAD + T - 1) / T, T, 0, stream>>>(ps, spc, pT);
  // per-edge clade einsum + mean/std
  gtg_edge<<<NED / 16, 32, 0, stream>>>(cmh, pT, rnb, cmr, out);
}
